// GPSRiskModel_9483287790248
// MI455X (gfx1250) — matrix operand order repaired and checked by reading, not compiled, not run on hardware
//
#include <hip/hip_runtime.h>
#include <hip/hip_bf16.h>

typedef __attribute__((ext_vector_type(16))) __bf16 v16bf;
typedef __attribute__((ext_vector_type(8)))  __bf16 v8bf;
typedef __attribute__((ext_vector_type(8)))  float  v8f;

#define NN   4096     // nodes
#define EE   262144   // edges
#define IND  64       // input feature dim
#define HD   128      // hidden
#define NH   4        // heads
#define DHD  32       // per-head dim
#define NL   2        // layers
#define OUTD 2        // output dim
#define NEG_SLOPE 0.2f

static __device__ __forceinline__ v8f wmma_bf16(v16bf a, v16bf b, v8f c) {
  return __builtin_amdgcn_wmma_f32_16x16x32_bf16(false, a, false, b, (short)0, c, false, false);
}

// ---------------- f32 -> bf16 convert ----------------
__global__ void k_convert(const float* __restrict__ in, __bf16* __restrict__ outp, int n) {
  int i = blockIdx.x * blockDim.x + threadIdx.x;
  if (i < n) outp[i] = (__bf16)in[i];
}

// ---------------- bf16 WMMA GEMM: C[M,N] = A[M,K] @ W[N,K]^T (+bias)(+relu) ----------
// One wave per 16x64 output tile: A-frag loaded once per k-step, 4 independent
// accumulators -> 4 back-to-back WMMAs with no D->C dependency chain.
// A-frag (16x32): lane row = lane&15; elem i: k = i + 16*(lane>>4)
// B-frag (32x16): lane col = lane&15; elem i: k = i + 16*(lane>>4)   (B[k][col] = W[col][k])
// C-frag (16x16): vgpr v: row = v + 8*(lane>>4), col = lane&15
__global__ void k_gemm(const __bf16* __restrict__ A, const __bf16* __restrict__ W,
                       const float* __restrict__ bias, float* __restrict__ C,
                       __bf16* __restrict__ Cbf, int M, int Nn, int K, int relu) {
  const int lane = threadIdx.x;
  const int hf   = lane >> 4;
  const int c    = lane & 15;
  const int m0   = blockIdx.x * 16;
  const int n0   = blockIdx.y * 64;
  const __bf16* Arow = A + (size_t)(m0 + c) * K;
  v8f acc0 = {}, acc1 = {}, acc2 = {}, acc3 = {};
  for (int k0 = 0; k0 < K; k0 += 32) {
    v8bf g0 = *(const v8bf*)(Arow + k0 + 16 * hf);
    v8bf g1 = *(const v8bf*)(Arow + k0 + 16 * hf + 8);
    v16bf a;
#pragma unroll
    for (int i = 0; i < 8; ++i) { a[i] = g0[i]; a[i + 8] = g1[i]; }
    v16bf b0 = *(const v16bf*)(W + (size_t)(n0 +  0 + c) * K + k0 + 16 * hf);
    v16bf b1 = *(const v16bf*)(W + (size_t)(n0 + 16 + c) * K + k0 + 16 * hf);
    v16bf b2 = *(const v16bf*)(W + (size_t)(n0 + 32 + c) * K + k0 + 16 * hf);
    v16bf b3 = *(const v16bf*)(W + (size_t)(n0 + 48 + c) * K + k0 + 16 * hf);
    acc0 = wmma_bf16(a, b0, acc0);
    acc1 = wmma_bf16(a, b1, acc1);
    acc2 = wmma_bf16(a, b2, acc2);
    acc3 = wmma_bf16(a, b3, acc3);
  }
  v8f* accs[4] = { &acc0, &acc1, &acc2, &acc3 };
#pragma unroll
  for (int j = 0; j < 4; ++j) {
    float bv = bias ? bias[n0 + 16 * j + c] : 0.0f;
    v8f acc = *accs[j];
#pragma unroll
    for (int v = 0; v < 8; ++v) {
      int row = m0 + v + 8 * hf;
      float val = acc[v] + bv;
      if (relu) val = fmaxf(val, 0.0f);
      size_t o = (size_t)row * Nn + n0 + 16 * j + c;
      C[o] = val;
      if (Cbf) Cbf[o] = (__bf16)val;
    }
  }
}

// ---------------- flash attention ----------------
// Block = 4 waves, each owning one 16-query tile (same head). The block
// cooperatively stages the 32-key K tile and a TRANSPOSED V tile into LDS with
// coalesced b128 loads; all WMMA fragments then come from 16/32-byte ds loads.
// qkv: bf16 [NN x 384] (q:0..127, k:128..255, v:256..383)
__global__ void k_attn(const __bf16* __restrict__ qkv, float* __restrict__ outp,
                       __bf16* __restrict__ out_bf) {
  __shared__ __attribute__((aligned(64))) __bf16 ldsK[32 * 32];   // [key][dh]
  __shared__ __attribute__((aligned(64))) __bf16 ldsVt[32 * 32];  // [dh][key]
  __shared__ __attribute__((aligned(64))) __bf16 ldsP[4][16 * 32];
  const int tid  = threadIdx.x;        // 0..127
  const int wave = tid >> 5;
  const int lane = tid & 31;
  const int hf   = lane >> 4;
  const int c    = lane & 15;
  const int q0   = blockIdx.x * 64 + wave * 16;
  const int head = blockIdx.y;
  const int S3H  = 3 * HD;
  const int qoff = head * DHD;
  const int koff = HD + head * DHD;
  const int voff = 2 * HD + head * DHD;
  const float sc = 0.17677669529663689f; // 1/sqrt(32)
  // staging role: key row + 8-wide dh group
  const int srow = tid >> 2;           // 0..31
  const int scol = (tid & 3) * 8;      // 0,8,16,24

  // Q fragment (A layout), rows q0..q0+15, K-dim = DHD = 32
  const __bf16* Qrow = qkv + (size_t)(q0 + c) * S3H + qoff;
  v8bf qg0 = *(const v8bf*)(Qrow + 16 * hf);
  v8bf qg1 = *(const v8bf*)(Qrow + 16 * hf + 8);
  v16bf qa;
#pragma unroll
  for (int i = 0; i < 8; ++i) { qa[i] = qg0[i]; qa[i + 8] = qg1[i]; }

  float m[8], l[8];
  v8f o0 = {}, o1 = {};
#pragma unroll
  for (int v = 0; v < 8; ++v) { m[v] = -3.0e38f; l[v] = 0.0f; }

  for (int key0 = 0; key0 < NN; key0 += 32) {
    // ---- cooperative stage: K row-major, V transposed ----
    const __bf16* Krow = qkv + (size_t)(key0 + srow) * S3H;
    v8bf kv = *(const v8bf*)(Krow + koff + scol);
    v8bf vv = *(const v8bf*)(Krow + voff + scol);
    *(v8bf*)(ldsK + srow * 32 + scol) = kv;
#pragma unroll
    for (int i = 0; i < 8; ++i) ldsVt[(scol + i) * 32 + srow] = vv[i];
    if (key0 + 32 < NN) {  // hide next tile's latency (global_prefetch_b8)
      const __bf16* Knext = qkv + (size_t)(key0 + 32 + srow) * S3H;
      __builtin_prefetch(Knext + koff + scol, 0, 0);
      __builtin_prefetch(Knext + voff + scol, 0, 0);
    }
    __syncthreads();

    // ---- scores: S[16x32] = Q @ K^T (two WMMA) ----
    v16bf kb0 = *(const v16bf*)(ldsK + c * 32 + 16 * hf);
    v16bf kb1 = *(const v16bf*)(ldsK + (16 + c) * 32 + 16 * hf);
    v8f z = {};
    v8f s0 = wmma_bf16(qa, kb0, z);
    v8f s1 = wmma_bf16(qa, kb1, z);

    // ---- online softmax (row r of C-frag lives across its 16-lane group) ----
#pragma unroll
    for (int v = 0; v < 8; ++v) {
      float t0 = s0[v] * sc, t1 = s1[v] * sc;
      float rm = fmaxf(t0, t1);
      rm = fmaxf(rm, __shfl_xor(rm, 1, 32));
      rm = fmaxf(rm, __shfl_xor(rm, 2, 32));
      rm = fmaxf(rm, __shfl_xor(rm, 4, 32));
      rm = fmaxf(rm, __shfl_xor(rm, 8, 32));
      float mn = fmaxf(m[v], rm);
      float f  = __expf(m[v] - mn);
      float p0 = __expf(t0 - mn);
      float p1 = __expf(t1 - mn);
      float rs = p0 + p1;
      rs += __shfl_xor(rs, 1, 32);
      rs += __shfl_xor(rs, 2, 32);
      rs += __shfl_xor(rs, 4, 32);
      rs += __shfl_xor(rs, 8, 32);
      l[v] = l[v] * f + rs;
      m[v] = mn;
      o0[v] *= f; o1[v] *= f;
      int row = v + 8 * hf;
      ldsP[wave][row * 32 + c]      = (__bf16)p0;
      ldsP[wave][row * 32 + 16 + c] = (__bf16)p1;
    }
    __syncthreads();

    // ---- O += P @ V : P re-fragmented from LDS, V frags from transposed LDS ----
    v8bf pg0 = *(const v8bf*)(ldsP[wave] + c * 32 + 16 * hf);
    v8bf pg1 = *(const v8bf*)(ldsP[wave] + c * 32 + 16 * hf + 8);
    v16bf pa;
#pragma unroll
    for (int i = 0; i < 8; ++i) { pa[i] = pg0[i]; pa[i + 8] = pg1[i]; }
    v16bf vb0 = *(const v16bf*)(ldsVt + c * 32 + 16 * hf);         // dh = c
    v16bf vb1 = *(const v16bf*)(ldsVt + (16 + c) * 32 + 16 * hf);  // dh = 16 + c
    o0 = wmma_bf16(pa, vb0, o0);
    o1 = wmma_bf16(pa, vb1, o1);
    __syncthreads();   // all LDS reads done before next stage overwrites
  }
#pragma unroll
  for (int v = 0; v < 8; ++v) {
    float inv = 1.0f / l[v];
    int row = q0 + v + 8 * hf;
    float a0 = o0[v] * inv, a1 = o1[v] * inv;
    size_t b0o = (size_t)row * HD + head * DHD + c;
    outp[b0o]        = a0;
    outp[b0o + 16]   = a1;
    out_bf[b0o]      = (__bf16)a0;
    out_bf[b0o + 16] = (__bf16)a1;
  }
}

// ---------------- GAT edge path ----------------
__global__ void k_node_alpha(const float* __restrict__ xp, const float* __restrict__ asrc,
                             const float* __restrict__ adst, float* __restrict__ al_s,
                             float* __restrict__ al_d) {
  int idx = blockIdx.x * blockDim.x + threadIdx.x;
  if (idx >= NN * NH) return;
  int n = idx >> 2, hh = idx & 3;
  const float* xr = xp + (size_t)n * HD + hh * DHD;
  float s = 0.0f, d = 0.0f;
#pragma unroll
  for (int i = 0; i < DHD; ++i) { s += xr[i] * asrc[hh * DHD + i]; d += xr[i] * adst[hh * DHD + i]; }
  al_s[idx] = s; al_d[idx] = d;
}

__global__ void k_seg_init(float* __restrict__ accum, float* __restrict__ denom,
                           unsigned* __restrict__ segmax) {
  int idx = blockIdx.x * blockDim.x + threadIdx.x;
  if (idx < NN * HD) accum[idx] = 0.0f;
  if (idx < NN * NH) { denom[idx] = 0.0f; segmax[idx] = 0u; }
}

static __device__ __forceinline__ unsigned f_enc(float x) {
  unsigned u = __float_as_uint(x);
  return (u & 0x80000000u) ? ~u : (u | 0x80000000u);
}
static __device__ __forceinline__ float f_dec(unsigned u) {
  return (u & 0x80000000u) ? __uint_as_float(u ^ 0x80000000u) : __uint_as_float(~u);
}

__global__ void k_edge_max(const int* __restrict__ ei, const float* __restrict__ al_s,
                           const float* __restrict__ al_d, unsigned* __restrict__ segmax) {
  int idx = blockIdx.x * blockDim.x + threadIdx.x;
  if (idx >= EE * NH) return;
  int e = idx >> 2, hh = idx & 3;
  int s = ei[e], d = ei[EE + e];
  float lg = al_s[s * NH + hh] + al_d[d * NH + hh];
  lg = (lg >= 0.0f) ? lg : NEG_SLOPE * lg;
  atomicMax(&segmax[d * NH + hh], f_enc(lg));
}

__global__ void k_seg_decode(const unsigned* __restrict__ segmax, float* __restrict__ seg_m) {
  int idx = blockIdx.x * blockDim.x + threadIdx.x;
  if (idx >= NN * NH) return;
  unsigned u = segmax[idx];
  seg_m[idx] = (u == 0u) ? 0.0f : f_dec(u);   // empty segment -> 0 (matches reference)
}

__global__ void k_edge_accum(const int* __restrict__ ei, const float* __restrict__ al_s,
                             const float* __restrict__ al_d, const float* __restrict__ seg_m,
                             const float* __restrict__ xp, float* __restrict__ denom,
                             float* __restrict__ accum) {
  int idx = blockIdx.x * blockDim.x + threadIdx.x;
  if (idx >= EE * NH) return;
  int e = idx >> 2, hh = idx & 3;
  int s = ei[e], d = ei[EE + e];
  float lg = al_s[s * NH + hh] + al_d[d * NH + hh];
  lg = (lg >= 0.0f) ? lg : NEG_SLOPE * lg;
  float el = __expf(lg - seg_m[d * NH + hh]);
  atomicAdd(&denom[d * NH + hh], el);
  const float* xs = xp + (size_t)s * HD + hh * DHD;
  float* ac = accum + (size_t)d * HD + hh * DHD;
#pragma unroll
  for (int i = 0; i < DHD; ++i) atomicAdd(&ac[i], el * xs[i]);
}

// ---------------- elementwise fusions ----------------
__global__ void k_gat_final(const float* __restrict__ accum, const float* __restrict__ denom,
                            const float* __restrict__ bgat, const float* __restrict__ h,
                            const float* __restrict__ g, const float* __restrict__ b,
                            float* __restrict__ h_loc) {
  int idx = blockIdx.x * blockDim.x + threadIdx.x;
  if (idx >= NN * HD) return;
  int n = idx >> 7, cc = idx & 127, hh = cc >> 5;
  const float bnS = rsqrtf(1.0f + 1e-5f);
  float val = accum[idx] / (denom[n * NH + hh] + 1e-16f) + bgat[cc] + h[idx];
  h_loc[idx] = val * (g[cc] * bnS) + b[cc];
}

__global__ void k_att_final(const float* __restrict__ proj, const float* __restrict__ h,
                            const float* __restrict__ g, const float* __restrict__ b,
                            float* __restrict__ h_att) {
  int idx = blockIdx.x * blockDim.x + threadIdx.x;
  if (idx >= NN * HD) return;
  int cc = idx & 127;
  const float bnS = rsqrtf(1.0f + 1e-5f);
  float val = proj[idx] + h[idx];
  h_att[idx] = val * (g[cc] * bnS) + b[cc];
}

__global__ void k_sum(const float* __restrict__ a, const float* __restrict__ b,
                      float* __restrict__ o, __bf16* __restrict__ obf) {
  int idx = blockIdx.x * blockDim.x + threadIdx.x;
  if (idx >= NN * HD) return;
  float v = a[idx] + b[idx];
  o[idx] = v; obf[idx] = (__bf16)v;
}

__global__ void k_layer_final(const float* __restrict__ osum, const float* __restrict__ t2,
                              const float* __restrict__ g, const float* __restrict__ b,
                              float* __restrict__ h, __bf16* __restrict__ h_bf) {
  int idx = blockIdx.x * blockDim.x + threadIdx.x;
  if (idx >= NN * HD) return;
  int cc = idx & 127;
  const float bnS = rsqrtf(1.0f + 1e-5f);
  float val = osum[idx] + t2[idx];
  val = val * (g[cc] * bnS) + b[cc];
  val = fmaxf(val + h[idx], 0.0f);
  h[idx] = val; h_bf[idx] = (__bf16)val;
}

__global__ void k_out(const float* __restrict__ h, const float* __restrict__ w,
                      const float* __restrict__ b, float* __restrict__ outp) {
  int idx = blockIdx.x * blockDim.x + threadIdx.x;
  if (idx >= NN * OUTD) return;
  int n = idx >> 1, o = idx & 1;
  const float* hr = h + (size_t)n * HD;
  const float* wr = w + (size_t)o * HD;
  float acc = b[o];
#pragma unroll 8
  for (int i = 0; i < HD; ++i) acc += hr[i] * wr[i];
  outp[idx] = acc;
}

extern "C" void kernel_launch(void* const* d_in, const int* in_sizes, int n_in,
                              void* d_out, int out_size, void* d_ws, size_t ws_size,
                              hipStream_t stream) {
  (void)in_sizes; (void)n_in; (void)out_size; (void)ws_size;
  const float* x     = (const float*)d_in[0];
  const int*   ei    = (const int*)d_in[1];
  const float* w_in  = (const float*)d_in[2];
  const float* b_in  = (const float*)d_in[3];
  const float* w_gat = (const float*)d_in[4];
  const float* att_s = (const float*)d_in[5];
  const float* att_d = (const float*)d_in[6];
  const float* b_gat = (const float*)d_in[7];
  const float* w_qkv = (const float*)d_in[8];
  const float* b_qkv = (const float*)d_in[9];
  const float* w_o   = (const float*)d_in[10];
  const float* b_o   = (const float*)d_in[11];
  const float* bn_g  = (const float*)d_in[12];
  const float* bn_b  = (const float*)d_in[13];
  const float* w1    = (const float*)d_in[14];
  const float* b1    = (const float*)d_in[15];
  const float* w2    = (const float*)d_in[16];
  const float* b2    = (const float*)d_in[17];
  const float* w_out = (const float*)d_in[18];
  const float* b_out = (const float*)d_in[19];
  float* outp = (float*)d_out;

  size_t off = 0;
  auto alloc = [&](size_t bytes) -> void* {
    void* p = (char*)d_ws + off;
    off += (bytes + 255) & ~(size_t)255;
    return p;
  };
  float*    h       = (float*)alloc((size_t)NN * HD * 4);
  __bf16*   h_bf    = (__bf16*)alloc((size_t)NN * HD * 2);
  __bf16*   x_bf    = (__bf16*)alloc((size_t)NN * IND * 2);
  __bf16*   w_in_bf = (__bf16*)alloc((size_t)HD * IND * 2);
  __bf16*   w_gat_bf= (__bf16*)alloc((size_t)NL * HD * HD * 2);
  __bf16*   w_qkv_bf= (__bf16*)alloc((size_t)NL * 3 * HD * HD * 2);
  __bf16*   w_o_bf  = (__bf16*)alloc((size_t)NL * HD * HD * 2);
  __bf16*   w1_bf   = (__bf16*)alloc((size_t)NL * 2 * HD * HD * 2);
  __bf16*   w2_bf   = (__bf16*)alloc((size_t)NL * HD * 2 * HD * 2);
  float*    xp      = (float*)alloc((size_t)NN * HD * 4);
  float*    al_s    = (float*)alloc((size_t)NN * NH * 4);
  float*    al_d    = (float*)alloc((size_t)NN * NH * 4);
  unsigned* segmax  = (unsigned*)alloc((size_t)NN * NH * 4);
  float*    seg_m   = (float*)alloc((size_t)NN * NH * 4);
  float*    denom   = (float*)alloc((size_t)NN * NH * 4);
  float*    accum   = (float*)alloc((size_t)NN * HD * 4);
  float*    h_loc   = (float*)alloc((size_t)NN * HD * 4);
  float*    qkv     = (float*)alloc((size_t)NN * 3 * HD * 4);
  __bf16*   qkv_bf  = (__bf16*)alloc((size_t)NN * 3 * HD * 2);
  float*    attn    = (float*)alloc((size_t)NN * HD * 4);
  __bf16*   attn_bf = (__bf16*)alloc((size_t)NN * HD * 2);
  float*    proj    = (float*)alloc((size_t)NN * HD * 4);
  float*    h_att   = (float*)alloc((size_t)NN * HD * 4);
  float*    osum    = (float*)alloc((size_t)NN * HD * 4);
  __bf16*   osum_bf = (__bf16*)alloc((size_t)NN * HD * 2);
  float*    t1      = (float*)alloc((size_t)NN * 2 * HD * 4);
  __bf16*   t1_bf   = (__bf16*)alloc((size_t)NN * 2 * HD * 2);
  float*    t2      = (float*)alloc((size_t)NN * HD * 4);

  auto cvt = [&](const float* s, __bf16* d, int n) {
    k_convert<<<(n + 255) / 256, 256, 0, stream>>>(s, d, n);
  };
  auto gemm = [&](const __bf16* A, const __bf16* W, const float* bias,
                  float* C, __bf16* Cbf, int M, int Nn, int K, int relu) {
    dim3 g(M / 16, Nn / 64);
    k_gemm<<<g, 32, 0, stream>>>(A, W, bias, C, Cbf, M, Nn, K, relu);
  };

  // weight + input conversion (tiny, once per call)
  cvt(x, x_bf, NN * IND);
  cvt(w_in, w_in_bf, HD * IND);
  cvt(w_gat, w_gat_bf, NL * HD * HD);
  cvt(w_qkv, w_qkv_bf, NL * 3 * HD * HD);
  cvt(w_o, w_o_bf, NL * HD * HD);
  cvt(w1, w1_bf, NL * 2 * HD * HD);
  cvt(w2, w2_bf, NL * HD * 2 * HD);

  // input projection: h = relu(x @ w_in^T + b_in)
  gemm(x_bf, w_in_bf, b_in, h, h_bf, NN, HD, IND, 1);

  const int EG  = (EE * NH + 255) / 256;
  const int NG  = (NN * HD + 255) / 256;
  const int NHG = (NN * NH + 255) / 256;

  for (int l = 0; l < NL; ++l) {
    // ---- local GAT branch ----
    gemm(h_bf, w_gat_bf + (size_t)l * HD * HD, nullptr, xp, nullptr, NN, HD, HD, 0);
    k_node_alpha<<<NHG, 256, 0, stream>>>(xp, att_s + l * NH * DHD, att_d + l * NH * DHD, al_s, al_d);
    k_seg_init<<<NG, 256, 0, stream>>>(accum, denom, segmax);
    k_edge_max<<<EG, 256, 0, stream>>>(ei, al_s, al_d, segmax);
    k_seg_decode<<<NHG, 256, 0, stream>>>(segmax, seg_m);
    k_edge_accum<<<EG, 256, 0, stream>>>(ei, al_s, al_d, seg_m, xp, denom, accum);
    k_gat_final<<<NG, 256, 0, stream>>>(accum, denom, b_gat + l * HD, h,
                                        bn_g + (l * 3 + 0) * HD, bn_b + (l * 3 + 0) * HD, h_loc);
    // ---- global attention branch ----
    gemm(h_bf, w_qkv_bf + (size_t)l * 3 * HD * HD, b_qkv + l * 3 * HD, qkv, qkv_bf, NN, 3 * HD, HD, 0);
    k_attn<<<dim3(NN / 64, NH), 128, 0, stream>>>(qkv_bf, attn, attn_bf);
    gemm(attn_bf, w_o_bf + (size_t)l * HD * HD, b_o + l * HD, proj, nullptr, NN, HD, HD, 0);
    k_att_final<<<NG, 256, 0, stream>>>(proj, h,
                                        bn_g + (l * 3 + 1) * HD, bn_b + (l * 3 + 1) * HD, h_att);
    // ---- combine + MLP ----
    k_sum<<<NG, 256, 0, stream>>>(h_loc, h_att, osum, osum_bf);
    gemm(osum_bf, w1_bf + (size_t)l * 2 * HD * HD, b1 + l * 2 * HD, t1, t1_bf, NN, 2 * HD, HD, 1);
    gemm(t1_bf, w2_bf + (size_t)l * HD * 2 * HD, b2 + l * HD, t2, nullptr, NN, HD, 2 * HD, 0);
    k_layer_final<<<NG, 256, 0, stream>>>(osum, t2,
                                          bn_g + (l * 3 + 2) * HD, bn_b + (l * 3 + 2) * HD, h, h_bf);
  }
  // final projection (tiny N=2: plain VALU dot)
  k_out<<<(NN * OUTD + 255) / 256, 256, 0, stream>>>(h, w_out, b_out, outp);
}
